// GaitPINN_37907381354720
// MI455X (gfx1250) — compile-verified, matrix operand order checked by reading, not run on hardware
//
#include <hip/hip_runtime.h>
#include <hip/hip_bf16.h>
#include <math.h>

// ---------------------------------------------------------------------------
// GaitPINN forward for MI455X (gfx1250, wave32, WMMA + async global->LDS DMA).
// Outputs (float32, concatenated): z0(1024*2), z_data(256*1024*2),
// z_traj_ode(256*1024*2), x_hat(256*1024*9), omega(1024).
// ---------------------------------------------------------------------------

typedef _Float16 half_t;
typedef __attribute__((ext_vector_type(16))) _Float16 v16h;
typedef __attribute__((ext_vector_type(8)))  float    v8f;

#define BATCH 1024
#define TLEN  256
#define CIN   9
#define HENC  128

#define OUT_Z0    0
#define OUT_ZDATA 2048
#define OUT_ZTRAJ 526336
#define OUT_XHAT  1050624
#define OUT_OMEGA 3409920

// f16 weight image in d_ws (halves): W1t [128][32] | W2t [128][128] | W3t [16][128]
#define WS_W1T 0
#define WS_W2T 4096
#define WS_W3T 20480
#define WS_TOTAL 22528   // halves (45056 bytes)

#define TWO_PI_F 6.283185307179586f

union FragU { unsigned int u[8]; v16h h; };

// ---- async global->LDS (CDNA5 GLOBAL_LOAD_ASYNC_TO_LDS_B128, ASYNCcnt) ----
#if defined(__has_builtin) && __has_builtin(__builtin_amdgcn_global_load_async_to_lds_b128)
#define HAVE_ASYNC_LDS 1
typedef int v4i __attribute__((vector_size(16)));
typedef __attribute__((address_space(1))) v4i glob_v4i;
typedef __attribute__((address_space(3))) v4i lds_v4i;
__device__ __forceinline__ void async_copy16(const void* g, void* l) {
  __builtin_amdgcn_global_load_async_to_lds_b128((glob_v4i*)g, (lds_v4i*)l, 0, 0);
}
__device__ __forceinline__ void async_wait0() {
#if __has_builtin(__builtin_amdgcn_s_wait_asynccnt)
  __builtin_amdgcn_s_wait_asynccnt(0);
#else
  asm volatile("s_wait_asynccnt 0x0" ::: "memory");
#endif
}
#else
#define HAVE_ASYNC_LDS 0
__device__ __forceinline__ void async_copy16(const void* g, void* l) {
  *(float4*)l = *(const float4*)g;
}
__device__ __forceinline__ void async_wait0() {}
#endif

// Load one 16x32 f16 WMMA operand fragment from an LDS buffer laid out
// row-major [rows][ldk] (rows = M for A, rows = N for B stored transposed).
// CDNA5 f16 A layout: lanes 0-15 -> M=row, K in {0..7,16..23};
// lanes 16-31 -> same M rows, K in {8..15,24..31}; each VGPR packs (k,k+1).
__device__ __forceinline__ v16h ldsFrag(const half_t* buf, int ldk, int r0,
                                        int kbase, int lane) {
  const int g = lane >> 4;
  const int r = r0 + (lane & 15);
  FragU f;
#pragma unroll
  for (int v = 0; v < 8; ++v) {
    const int k = kbase + ((v < 4) ? 0 : 16) + g * 8 + 2 * (v & 3);
    f.u[v] = *reinterpret_cast<const unsigned int*>(buf + r * ldk + k);
  }
  return f.h;
}

__device__ __forceinline__ v8f wmma_f16(v16h a, v16h b, v8f c) {
  return __builtin_amdgcn_wmma_f32_16x16x32_f16(false, a, false, b,
                                                (short)0, c, false, false);
}

// ---------------------------------------------------------------------------
// Kernel 0: one-time f32 -> f16 weight transpose/pad into d_ws.
// ---------------------------------------------------------------------------
__global__ __launch_bounds__(256) void gaitpinn_prep_kernel(
    const float* __restrict__ enc_w1, const float* __restrict__ enc_w2,
    const float* __restrict__ enc_w3, half_t* __restrict__ ws) {
  const int idx = blockIdx.x * 256 + threadIdx.x;
  if (idx >= WS_TOTAL) return;
  float v;
  if (idx < WS_W2T) {                      // W1t [n][k], k padded 9->32
    const int n = idx >> 5, k = idx & 31;
    v = (k < CIN) ? enc_w1[k * HENC + n] : 0.f;
  } else if (idx < WS_W3T) {               // W2t [n][k]
    const int j = idx - WS_W2T;
    const int n = j >> 7, k = j & 127;
    v = enc_w2[k * HENC + n];
  } else {                                 // W3t [n][k], n padded 2->16
    const int j = idx - WS_W3T;
    const int n = j >> 7, k = j & 127;
    v = (n < 2) ? enc_w3[k * 2 + n] : 0.f;
  }
  ws[idx] = (half_t)v;
}

// ---------------------------------------------------------------------------
// Kernel 1: one workgroup per sample b. 256 threads (8 waves).
// Fused encoder (9->128->128->2) via WMMA + decoder + omega head.
// ---------------------------------------------------------------------------
__global__ __launch_bounds__(256) void gaitpinn_encoder_kernel(
    const float* __restrict__ x, const half_t* __restrict__ wsw,
    const float* __restrict__ enc_b1, const float* __restrict__ enc_b2,
    const float* __restrict__ enc_b3,
    const float* __restrict__ dec_w,  const float* __restrict__ dec_b,
    const float* __restrict__ om_w1,  const float* __restrict__ om_b1,
    const float* __restrict__ om_w2,  const float* __restrict__ om_b2,
    float* __restrict__ out) {
  __shared__ half_t WH[WS_TOTAL];       // W1t|W2t|W3t f16 image     (44 KB)
  __shared__ float  Xraw[TLEN * CIN];   // raw x tile (async DMA)    (9 KB)
  __shared__ half_t Xh[TLEN * 32];      // activations, K pad 9->32  (16 KB)
  __shared__ half_t H1[TLEN * HENC];    // h1, then in-place h2      (64 KB)
  __shared__ float  B1s[HENC], B2s[HENC], B3p[16];
  __shared__ float  Z[TLEN * 2];
  __shared__ float  zsum[2];

  const int tid  = threadIdx.x;
  const int b    = blockIdx.x;
  const int wave = tid >> 5;
  const int lane = tid & 31;

  const half_t* W1t = WH + WS_W1T;
  const half_t* W2t = WH + WS_W2T;
  const half_t* W3t = WH + WS_W3T;

  // ---- async DMA: f16 weight image + raw x tile, straight into LDS ----
  for (int i = tid; i < WS_TOTAL / 8; i += 256)        // 2816 x 16B
    async_copy16(wsw + i * 8, WH + i * 8);
  {
    const float* xb = x + (size_t)b * (TLEN * CIN);    // 9216 B, 16B aligned
    for (int i = tid; i < (TLEN * CIN) / 4; i += 256)  // 576 x 16B
      async_copy16(xb + i * 4, Xraw + i * 4);
  }
  // small tables via regular loads meanwhile
  if (tid < HENC) { B1s[tid] = enc_b1[tid]; B2s[tid] = enc_b2[tid]; }
  if (tid < 16)   { B3p[tid] = (tid < 2) ? enc_b3[tid] : 0.f; }
  if (tid < 2)    { zsum[tid] = 0.f; }
  async_wait0();
  __syncthreads();

  // ---- pad/convert x tile: Xraw (f32, 256x9) -> Xh (f16, 256x32) ----
  for (int i = tid; i < TLEN * 32; i += 256) {
    const int r = i >> 5, k = i & 31;
    Xh[i] = (k < CIN) ? (half_t)Xraw[r * CIN + k] : (half_t)0.f;
  }
  __syncthreads();

  // ---- layer 1: h1 = relu(X @ W1 + b1), X is 256x9 (K padded to 32) ----
#pragma unroll
  for (int mt = 0; mt < 2; ++mt) {
    const int m0 = wave * 32 + mt * 16;
    const v16h A = ldsFrag(Xh, 32, m0, 0, lane);
    for (int nt = 0; nt < 8; ++nt) {
      const float bn = B1s[nt * 16 + (lane & 15)];
      v8f c;
#pragma unroll
      for (int v = 0; v < 8; ++v) c[v] = bn;
      const v16h B = ldsFrag(W1t, 32, nt * 16, 0, lane);
      c = wmma_f16(A, B, c);
      const int col = nt * 16 + (lane & 15);
#pragma unroll
      for (int v = 0; v < 8; ++v) {
        const int row = m0 + v + 8 * (lane >> 4);
        H1[row * HENC + col] = (half_t)fmaxf(c[v], 0.f);
      }
    }
  }
  __syncthreads();

  // ---- layer 2: h2 = relu(h1 @ W2 + b2), in place on H1 (wave-local rows) ----
#pragma unroll
  for (int mt = 0; mt < 2; ++mt) {
    const int m0 = wave * 32 + mt * 16;
    v8f acc[8];
#pragma unroll
    for (int nt = 0; nt < 8; ++nt) {
      const float bn = B2s[nt * 16 + (lane & 15)];
#pragma unroll
      for (int v = 0; v < 8; ++v) acc[nt][v] = bn;
    }
    for (int ks = 0; ks < 4; ++ks) {
      const v16h A = ldsFrag(H1, HENC, m0, ks * 32, lane);
#pragma unroll
      for (int nt = 0; nt < 8; ++nt) {
        const v16h B = ldsFrag(W2t, HENC, nt * 16, ks * 32, lane);
        acc[nt] = wmma_f16(A, B, acc[nt]);
      }
    }
#pragma unroll
    for (int nt = 0; nt < 8; ++nt) {
      const int col = nt * 16 + (lane & 15);
#pragma unroll
      for (int v = 0; v < 8; ++v) {
        const int row = m0 + v + 8 * (lane >> 4);
        H1[row * HENC + col] = (half_t)fmaxf(acc[nt][v], 0.f);
      }
    }
  }
  __syncthreads();

  // ---- layer 3: z = h2 @ W3 + b3 (N padded 2->16) ----
#pragma unroll
  for (int mt = 0; mt < 2; ++mt) {
    const int m0 = wave * 32 + mt * 16;
    const float bn = ((lane & 15) < 2) ? B3p[lane & 15] : 0.f;
    v8f c;
#pragma unroll
    for (int v = 0; v < 8; ++v) c[v] = bn;
    for (int ks = 0; ks < 4; ++ks) {
      const v16h A = ldsFrag(H1, HENC, m0, ks * 32, lane);
      const v16h B = ldsFrag(W3t, HENC, 0, ks * 32, lane);
      c = wmma_f16(A, B, c);
    }
    const int n = lane & 15;
    if (n < 2) {
#pragma unroll
      for (int v = 0; v < 8; ++v) {
        const int row = m0 + v + 8 * (lane >> 4);
        Z[row * 2 + n] = c[v];
      }
    }
  }
  __syncthreads();

  // ---- epilogue: z_data / z0 / x_hat (decoder) / z_mean accumulation ----
  {
    const int t = tid;  // 256 threads == 256 time steps
    const float zx = Z[t * 2 + 0];
    const float zy = Z[t * 2 + 1];
    float* zd = out + OUT_ZDATA;
    zd[(t * BATCH + b) * 2 + 0] = zx;
    zd[(t * BATCH + b) * 2 + 1] = zy;
    if (t == 0) {
      out[OUT_Z0 + b * 2 + 0] = zx;
      out[OUT_Z0 + b * 2 + 1] = zy;
    }
    float* xh = out + OUT_XHAT + (size_t)(t * BATCH + b) * 9;
#pragma unroll
    for (int j = 0; j < 9; ++j)
      xh[j] = dec_b[j] + zx * dec_w[j] + zy * dec_w[9 + j];
    atomicAdd(&zsum[0], zx);
    atomicAdd(&zsum[1], zy);
  }
  __syncthreads();

  // ---- omega head (single thread: 2->16 relu -> 1 -> sigmoid) ----
  if (tid == 0) {
    const float zm0 = zsum[0] * (1.0f / TLEN);
    const float zm1 = zsum[1] * (1.0f / TLEN);
    float oraw = om_b2[0];
#pragma unroll
    for (int i = 0; i < 16; ++i) {
      const float h = fmaxf(zm0 * om_w1[i] + zm1 * om_w1[16 + i] + om_b1[i], 0.f);
      oraw += h * om_w2[i];
    }
    const float sig = 1.0f / (1.0f + __expf(-oraw));
    out[OUT_OMEGA + b] = (sig * 2.5f + 0.5f) * TWO_PI_F;
  }
}

// ---------------------------------------------------------------------------
// Kernel 2: RK4 (3/8 rule) Hopf + residual-MLP ODE. One wave32 per sample.
// Lane j owns hidden units j and j+32; outputs reduced with shfl_xor butterfly
// (sum ends up broadcast, so all lanes carry identical state).
// ---------------------------------------------------------------------------
__global__ __launch_bounds__(256) void gaitpinn_ode_kernel(
    const float* __restrict__ ode_w1, const float* __restrict__ ode_b1,
    const float* __restrict__ ode_w2, const float* __restrict__ ode_b2,
    const float* __restrict__ log_alpha, const float* __restrict__ log_epsilon,
    const float* __restrict__ cx_p, const float* __restrict__ cy_p,
    float* __restrict__ out) {
  const int lane = threadIdx.x & 31;
  const int wave = threadIdx.x >> 5;
  const int b    = blockIdx.x * 8 + wave;

  const int ja = lane, jb = lane + 32;
  const float w10a = ode_w1[ja],      w10b = ode_w1[jb];
  const float w11a = ode_w1[64 + ja], w11b = ode_w1[64 + jb];
  const float b1a  = ode_b1[ja],      b1b  = ode_b1[jb];
  const float w20a = ode_w2[ja * 2],     w21a = ode_w2[ja * 2 + 1];
  const float w20b = ode_w2[jb * 2],     w21b = ode_w2[jb * 2 + 1];
  const float ob0 = ode_b2[0], ob1 = ode_b2[1];

  const float alpha = __expf(*log_alpha);
  const float eps   = __expf(*log_epsilon);
  const float cx = *cx_p, cy = *cy_p;
  const float omega = out[OUT_OMEGA + b];
  const float dt = 3.0f / (float)(TLEN - 1);

  float zx = out[OUT_Z0 + b * 2 + 0];
  float zy = out[OUT_Z0 + b * 2 + 1];
  float* traj = out + OUT_ZTRAJ;
  if (lane == 0) {
    traj[(0 * BATCH + b) * 2 + 0] = zx;
    traj[(0 * BATCH + b) * 2 + 1] = zy;
  }

  auto ode_f = [&](float px, float py, float& dx, float& dy) {
    const float xc = px - cx, yc = py - cy;
    const float r2 = xc * xc + yc * yc + 1e-8f;
    const float rad = alpha * (1.0f - r2);
    const float hx = rad * xc - omega * yc;
    const float hy = rad * yc + omega * xc;
    const float ha = tanhf(px * w10a + py * w11a + b1a);
    const float hb = tanhf(px * w10b + py * w11b + b1b);
    float r0 = ha * w20a + hb * w20b;
    float r1 = ha * w21a + hb * w21b;
#pragma unroll
    for (int m = 16; m >= 1; m >>= 1) {
      r0 += __shfl_xor(r0, m, 32);
      r1 += __shfl_xor(r1, m, 32);
    }
    dx = fminf(fmaxf(hx + eps * (r0 + ob0), -20.0f), 20.0f);
    dy = fminf(fmaxf(hy + eps * (r1 + ob1), -20.0f), 20.0f);
  };

  for (int t = 0; t < TLEN - 1; ++t) {
    float k1x, k1y, k2x, k2y, k3x, k3y, k4x, k4y;
    ode_f(zx, zy, k1x, k1y);
    ode_f(zx + dt * k1x * (1.0f / 3.0f), zy + dt * k1y * (1.0f / 3.0f), k2x, k2y);
    ode_f(zx + dt * (k2x - k1x * (1.0f / 3.0f)),
          zy + dt * (k2y - k1y * (1.0f / 3.0f)), k3x, k3y);
    ode_f(zx + dt * (k1x - k2x + k3x), zy + dt * (k1y - k2y + k3y), k4x, k4y);
    zx += dt * 0.125f * (k1x + 3.0f * (k2x + k3x) + k4x);
    zy += dt * 0.125f * (k1y + 3.0f * (k2y + k3y) + k4y);
    if (lane == 0) {
      traj[((t + 1) * BATCH + b) * 2 + 0] = zx;
      traj[((t + 1) * BATCH + b) * 2 + 1] = zy;
    }
  }
}

extern "C" void kernel_launch(void* const* d_in, const int* in_sizes, int n_in,
                              void* d_out, int out_size, void* d_ws, size_t ws_size,
                              hipStream_t stream) {
  const float* x      = (const float*)d_in[0];
  const float* enc_w1 = (const float*)d_in[1];
  const float* enc_b1 = (const float*)d_in[2];
  const float* enc_w2 = (const float*)d_in[3];
  const float* enc_b2 = (const float*)d_in[4];
  const float* enc_w3 = (const float*)d_in[5];
  const float* enc_b3 = (const float*)d_in[6];
  const float* ode_w1 = (const float*)d_in[7];
  const float* ode_b1 = (const float*)d_in[8];
  const float* ode_w2 = (const float*)d_in[9];
  const float* ode_b2 = (const float*)d_in[10];
  const float* dec_w  = (const float*)d_in[11];
  const float* dec_b  = (const float*)d_in[12];
  const float* om_w1  = (const float*)d_in[13];
  const float* om_b1  = (const float*)d_in[14];
  const float* om_w2  = (const float*)d_in[15];
  const float* om_b2  = (const float*)d_in[16];
  const float* log_alpha   = (const float*)d_in[17];
  // d_in[18] = log_omega (unused by the forward pass)
  const float* log_epsilon = (const float*)d_in[19];
  const float* cx = (const float*)d_in[20];
  const float* cy = (const float*)d_in[21];
  float* out = (float*)d_out;
  half_t* ws = (half_t*)d_ws;

  hipLaunchKernelGGL(gaitpinn_prep_kernel, dim3((WS_TOTAL + 255) / 256),
                     dim3(256), 0, stream, enc_w1, enc_w2, enc_w3, ws);
  hipLaunchKernelGGL(gaitpinn_encoder_kernel, dim3(BATCH), dim3(256), 0, stream,
                     x, ws, enc_b1, enc_b2, enc_b3,
                     dec_w, dec_b, om_w1, om_b1, om_w2, om_b2, out);
  hipLaunchKernelGGL(gaitpinn_ode_kernel, dim3(BATCH / 8), dim3(256), 0, stream,
                     ode_w1, ode_b1, ode_w2, ode_b2,
                     log_alpha, log_epsilon, cx, cy, out);
}
